// GATv2Convolution_72911364817017
// MI455X (gfx1250) — matrix operand order repaired and checked by reading, not compiled, not run on hardware
//
#include <hip/hip_runtime.h>

// ---------------------------------------------------------------------------
// GATv2 (2 layers) for MI455X / gfx1250, wave32 + WMMA bf16 matrix cores.
//  * Dense transforms: v_wmma_f32_16x16x32_bf16 with f32 accumulation.
//    All f32->bf16 conversion is hoisted into one-time pack kernels; the GEMM
//    inner loop is pure b128 fragment loads + WMMA.
//  * Edge phase: wave-per-edge, lane-per-channel coalesced gathers and
//    GLOBAL_ATOMIC_ADD_F32 scatters. All per-layer hot arrays (< 110 MB)
//    stay resident in the 192 MB L2.
// ---------------------------------------------------------------------------

typedef __attribute__((ext_vector_type(16))) __bf16 v16bf;
typedef __attribute__((ext_vector_type(8)))  float  v8f;

#define DEVFN static __device__ __forceinline__

DEVFN float leaky02(float x) { return x > 0.f ? x : 0.2f * x; }

// float -> bf16 round-to-nearest-even via integer ops (no dependence on
// __bf16 conversion lowering).
DEVFN __bf16 f2bf(float f) {
    unsigned u = __float_as_uint(f);
    u += 0x7FFFu + ((u >> 16) & 1u);
    unsigned short h = (unsigned short)(u >> 16);
    __bf16 r;
    __builtin_memcpy(&r, &h, sizeof(r));
    return r;
}

// Order-preserving float<->uint mapping so segment-max can use atomicMax(u32).
DEVFN unsigned ford(float f) {
    unsigned u = __float_as_uint(f);
    return (u & 0x80000000u) ? ~u : (u | 0x80000000u);
}
DEVFN float fdec(unsigned o) {
    unsigned u = (o & 0x80000000u) ? (o ^ 0x80000000u) : ~o;
    return __uint_as_float(u);
}
#define ORD_NEG_INF 0x007FFFFFu  // ford(-inf)

// ---------------------------------------------------------------------------
// One-time f32 -> bf16 row-major pack (vectorized; total must be mult. of 4).
// ---------------------------------------------------------------------------
__global__ __launch_bounds__(256) void pack_bf16(
    const float* __restrict__ in, __bf16* __restrict__ out, long total)
{
    const long i = ((long)blockIdx.x * blockDim.x + threadIdx.x) * 4;
    if (i >= total) return;
    const float4 v = *reinterpret_cast<const float4*>(in + i);
    __bf16 o[4] = { f2bf(v.x), f2bf(v.y), f2bf(v.z), f2bf(v.w) };
    __builtin_memcpy(out + i, o, 8);
}

// ---------------------------------------------------------------------------
// One-time pack of W[K, COUT] (f32) into bf16 B-fragments laid out exactly as
// the contiguous-k order (k = 16*half + e) that the GEMM's A loads also use:
//   frag[((kt*NT + nt)*32 + lane)*16 + e] = W[kt*32 + 16*(lane>=16) + e][nt*16 + (lane&15)]
// so each lane's fragment is 32 contiguous bytes.
// ---------------------------------------------------------------------------
__global__ __launch_bounds__(256) void pack_w_frags(
    const float* __restrict__ W, __bf16* __restrict__ frag, int K, int COUT)
{
    const int idx = blockIdx.x * blockDim.x + threadIdx.x;
    if (idx >= K * COUT) return;
    const int NT   = COUT / 16;
    const int e    = idx & 15;
    const int lane = (idx >> 4) & 31;
    const int rem  = idx >> 9;            // kt*NT + nt
    const int nt   = rem % NT;
    const int kt   = rem / NT;
    const int k    = kt * 32 + ((lane >> 4) * 16) + e;
    const int n    = nt * 16 + (lane & 15);
    frag[idx] = f2bf(W[(size_t)k * COUT + n]);
}

// ---------------------------------------------------------------------------
// Y[nrows, COUT] = Xbf[nrows, K] @ W + bias   (WMMA bf16, f32 accum)
// One wave32 per 16-row tile; wave computes full COUT width.
// Inner loop: 2x b128 (A) + 2x b128 (B) + v_wmma per tile-MAC.
// ---------------------------------------------------------------------------
template <int K, int COUT>
__global__ __launch_bounds__(128) void gemm_bias_wmma(
    const __bf16* __restrict__ Xbf, const __bf16* __restrict__ Wfrag,
    const float* __restrict__ bias, float* __restrict__ Y, int nrows)
{
    constexpr int NT = COUT / 16;
    const int lane = threadIdx.x & 31;
    const int wave = threadIdx.x >> 5;
    const int rowBase = (blockIdx.x * 4 + wave) * 16;
    if (rowBase >= nrows) return;

    const int half = lane >> 4;     // A: lanes 0-15 hold K 0-15; 16-31 hold 16-31
    const int aK0  = half * 16;
    const int n    = lane & 15;     // B/C/D column

    int row = rowBase + (lane & 15);
    if (row >= nrows) row = nrows - 1;   // clamp loads; stores guarded

    v8f acc[NT] = {};

#pragma unroll
    for (int kt = 0; kt < K / 32; ++kt) {
        const __bf16* xrow = Xbf + (size_t)row * K + kt * 32 + aK0;
        v16bf A;
        reinterpret_cast<uint4*>(&A)[0] = *reinterpret_cast<const uint4*>(xrow);       // K +0..7
        reinterpret_cast<uint4*>(&A)[1] = *reinterpret_cast<const uint4*>(xrow + 8);   // K +8..15

#pragma unroll
        for (int t = 0; t < NT; ++t) {
            const __bf16* bp = Wfrag + (((size_t)kt * NT + t) * 32 + lane) * 16;
            v16bf B;
            reinterpret_cast<uint4*>(&B)[0] = *reinterpret_cast<const uint4*>(bp);
            reinterpret_cast<uint4*>(&B)[1] = *reinterpret_cast<const uint4*>(bp + 8);
            acc[t] = __builtin_amdgcn_wmma_f32_16x16x32_bf16(
                false, A, false, B, (short)0, acc[t], false, false);
        }
    }

    // C/D layout: VGPR v -> row v (lanes 0-15) or v+8 (lanes 16-31), col = lane&15
    const int rOff = half * 8;
#pragma unroll
    for (int t = 0; t < NT; ++t) {
        const float bn = bias[t * 16 + n];
#pragma unroll
        for (int v = 0; v < 8; ++v) {
            const int r = rowBase + rOff + v;
            if (r < nrows) Y[(size_t)r * COUT + t * 16 + n] = acc[t][v] + bn;
        }
    }
}

// ---------------------------------------------------------------------------
// Init per-layer reduction buffers: m = -inf (ordinal), s = 0, acc = 0.
// ---------------------------------------------------------------------------
__global__ void init_buffers(unsigned* __restrict__ mord, float* __restrict__ ssum,
                             float* __restrict__ acc, int N, int C)
{
    const int i = blockIdx.x * blockDim.x + threadIdx.x;
    if (i < N) { mord[i] = ORD_NEG_INF; ssum[i] = 0.f; }
    const int total = N * C;
    for (int j = i; j < total; j += gridDim.x * blockDim.x) acc[j] = 0.f;
}

// ---------------------------------------------------------------------------
// Pass 1: per-edge logit e = att . leaky_relu(xl[src] + xr[dst]); segment max.
// One wave per edge (ids >= E are self loops). Lane-per-channel -> coalesced.
// ---------------------------------------------------------------------------
__global__ __launch_bounds__(256) void edge_logits(
    const int* __restrict__ src, const int* __restrict__ dst,
    const float* __restrict__ xl, const float* __restrict__ xr,
    const float* __restrict__ att, float* __restrict__ ebuf,
    unsigned* __restrict__ mord, int E, int N, int C)
{
    const int id   = (blockIdx.x * blockDim.x + threadIdx.x) >> 5;
    const int lane = threadIdx.x & 31;
    if (id >= E + N) return;
    int s, d;
    if (id < E) { s = src[id]; d = dst[id]; } else { s = d = id - E; }

    float a = 0.f;
    for (int c = lane; c < C; c += 32)
        a += att[c] * leaky02(xl[(size_t)s * C + c] + xr[(size_t)d * C + c]);
#pragma unroll
    for (int off = 16; off > 0; off >>= 1) a += __shfl_down(a, off, 32);

    if (lane == 0) {
        ebuf[id] = a;
        atomicMax(&mord[d], ford(a));
    }
}

// ---------------------------------------------------------------------------
// Pass 2: ex = exp(e - m[dst]); s[dst] += ex. One thread per edge.
// ---------------------------------------------------------------------------
__global__ __launch_bounds__(256) void edge_exp(
    const int* __restrict__ dst, float* __restrict__ ebuf,
    const unsigned* __restrict__ mord, float* __restrict__ ssum, int E, int N)
{
    const int id = blockIdx.x * blockDim.x + threadIdx.x;
    if (id >= E + N) return;
    const int d = (id < E) ? dst[id] : (id - E);
    const float ex = __expf(ebuf[id] - fdec(mord[d]));
    ebuf[id] = ex;
    atomicAdd(&ssum[d], ex);
}

// ---------------------------------------------------------------------------
// Pass 3: acc[dst] += (ex / s[dst]) * xl[src]. Wave per edge, coalesced
// GLOBAL_ATOMIC_ADD_F32; acc stays L2-resident.
// ---------------------------------------------------------------------------
__global__ __launch_bounds__(256) void edge_scatter(
    const int* __restrict__ src, const int* __restrict__ dst,
    const float* __restrict__ xl, const float* __restrict__ ebuf,
    const float* __restrict__ ssum, float* __restrict__ acc, int E, int N, int C)
{
    const int id   = (blockIdx.x * blockDim.x + threadIdx.x) >> 5;
    const int lane = threadIdx.x & 31;
    if (id >= E + N) return;
    int s, d;
    if (id < E) { s = src[id]; d = dst[id]; } else { s = d = id - E; }

    const float alpha = ebuf[id] / (ssum[d] + 1e-16f);
    for (int c = lane; c < C; c += 32)
        atomicAdd(&acc[(size_t)d * C + c], alpha * xl[(size_t)s * C + c]);
}

// ---------------------------------------------------------------------------
// out = (acc + bias) with optional ReLU (layer 1).
// ---------------------------------------------------------------------------
__global__ __launch_bounds__(256) void bias_act(
    const float* __restrict__ acc, const float* __restrict__ bias,
    float* __restrict__ out, int N, int C, int do_relu)
{
    const int i = blockIdx.x * blockDim.x + threadIdx.x;
    if (i >= N * C) return;
    float v = acc[i] + bias[i % C];
    out[i] = do_relu ? fmaxf(v, 0.f) : v;
}

// ---------------------------------------------------------------------------
// Host-side orchestration (all launches on `stream`; scratch carved from d_ws).
// ---------------------------------------------------------------------------
extern "C" void kernel_launch(void* const* d_in, const int* in_sizes, int n_in,
                              void* d_out, int out_size, void* d_ws, size_t ws_size,
                              hipStream_t stream)
{
    constexpr int F_IN = 128, HID = 64, N_CLS = 32;

    const float* x     = (const float*)d_in[0];
    const int*   ei    = (const int*)d_in[1];
    const float* W1l   = (const float*)d_in[2];
    const float* b1l   = (const float*)d_in[3];
    const float* W1r   = (const float*)d_in[4];
    const float* b1r   = (const float*)d_in[5];
    const float* att1  = (const float*)d_in[6];
    const float* bias1 = (const float*)d_in[7];
    const float* W2l   = (const float*)d_in[8];
    const float* b2l   = (const float*)d_in[9];
    const float* W2r   = (const float*)d_in[10];
    const float* b2r   = (const float*)d_in[11];
    const float* att2  = (const float*)d_in[12];
    const float* bias2 = (const float*)d_in[13];

    const int N = in_sizes[0] / F_IN;
    const int E = in_sizes[1] / 2;
    const int* src = ei;
    const int* dst = ei + E;

    // Workspace layout (256B-aligned slabs).
    char* p = (char*)d_ws;
    auto take = [&](size_t bytes) { char* r = p; p += (bytes + 255) & ~size_t(255); return r; };
    float*    xl1  = (float*)take((size_t)N * HID * sizeof(float));
    float*    xr1  = (float*)take((size_t)N * HID * sizeof(float));
    float*    acc1 = (float*)take((size_t)N * HID * sizeof(float));
    float*    h    = (float*)take((size_t)N * HID * sizeof(float));
    float*    ebuf = (float*)take((size_t)(E + N) * sizeof(float));
    unsigned* mord = (unsigned*)take((size_t)N * sizeof(unsigned));
    float*    ssum = (float*)take((size_t)N * sizeof(float));
    __bf16*   xbf  = (__bf16*)take((size_t)N * F_IN * sizeof(__bf16));
    __bf16*   hbf  = (__bf16*)take((size_t)N * HID * sizeof(__bf16));
    __bf16*   wf1l = (__bf16*)take((size_t)F_IN * HID * sizeof(__bf16));
    __bf16*   wf1r = (__bf16*)take((size_t)F_IN * HID * sizeof(__bf16));
    __bf16*   wf2l = (__bf16*)take((size_t)HID * N_CLS * sizeof(__bf16));
    __bf16*   wf2r = (__bf16*)take((size_t)HID * N_CLS * sizeof(__bf16));
    // Layer-2 transforms reuse layer-1 slabs (free after layer-1 scatter).
    float* xl2 = xl1;
    float* xr2 = xr1;
    float* out = (float*)d_out;  // layer-2 accumulator + final output

    const int items   = E + N;
    const int nodeBlk = (N + 255) / 256;
    const int waveBlk = (items + 7) / 8;      // 8 waves of 32 per 256-thread block
    const int thrBlk  = (items + 255) / 256;
    const int gemmBlk = (N + 63) / 64;        // 4 waves x 16 rows per block

    // ---------------- Layer 1 (C = 64) ----------------
    init_buffers<<<nodeBlk, 256, 0, stream>>>(mord, ssum, acc1, N, HID);
    {
        const long xTot = (long)N * F_IN;
        pack_bf16   <<<(int)((xTot / 4 + 255) / 256), 256, 0, stream>>>(x, xbf, xTot);
        pack_w_frags<<<(F_IN * HID + 255) / 256, 256, 0, stream>>>(W1l, wf1l, F_IN, HID);
        pack_w_frags<<<(F_IN * HID + 255) / 256, 256, 0, stream>>>(W1r, wf1r, F_IN, HID);
        gemm_bias_wmma<F_IN, HID><<<gemmBlk, 128, 0, stream>>>(xbf, wf1l, b1l, xl1, N);
        gemm_bias_wmma<F_IN, HID><<<gemmBlk, 128, 0, stream>>>(xbf, wf1r, b1r, xr1, N);
    }
    edge_logits <<<waveBlk, 256, 0, stream>>>(src, dst, xl1, xr1, att1, ebuf, mord, E, N, HID);
    edge_exp    <<<thrBlk, 256, 0, stream>>>(dst, ebuf, mord, ssum, E, N);
    edge_scatter<<<waveBlk, 256, 0, stream>>>(src, dst, xl1, ebuf, ssum, acc1, E, N, HID);
    bias_act    <<<(N * HID + 255) / 256, 256, 0, stream>>>(acc1, bias1, h, N, HID, 1);

    // ---------------- Layer 2 (C = 32) ----------------
    init_buffers<<<nodeBlk, 256, 0, stream>>>(mord, ssum, out, N, N_CLS);
    {
        const long hTot = (long)N * HID;
        pack_bf16   <<<(int)((hTot / 4 + 255) / 256), 256, 0, stream>>>(h, hbf, hTot);
        pack_w_frags<<<(HID * N_CLS + 255) / 256, 256, 0, stream>>>(W2l, wf2l, HID, N_CLS);
        pack_w_frags<<<(HID * N_CLS + 255) / 256, 256, 0, stream>>>(W2r, wf2r, HID, N_CLS);
        gemm_bias_wmma<HID, N_CLS><<<gemmBlk, 128, 0, stream>>>(hbf, wf2l, b2l, xl2, N);
        gemm_bias_wmma<HID, N_CLS><<<gemmBlk, 128, 0, stream>>>(hbf, wf2r, b2r, xr2, N);
    }
    edge_logits <<<waveBlk, 256, 0, stream>>>(src, dst, xl2, xr2, att2, ebuf, mord, E, N, N_CLS);
    edge_exp    <<<thrBlk, 256, 0, stream>>>(dst, ebuf, mord, ssum, E, N);
    edge_scatter<<<waveBlk, 256, 0, stream>>>(src, dst, xl2, ebuf, ssum, out, E, N, N_CLS);
    bias_act    <<<(N * N_CLS + 255) / 256, 256, 0, stream>>>(out, bias2, out, N, N_CLS, 0);
}
